// HyperbolicGraphConvolution_81887846466065
// MI455X (gfx1250) — matrix operand order repaired and checked by reading, compiled, not run on hardware
//
#include <hip/hip_runtime.h>
#include <math.h>

typedef __bf16 bf16;
typedef __attribute__((ext_vector_type(16))) __bf16 v16bf;
typedef __attribute__((ext_vector_type(8)))  __bf16 bf16x8;
typedef __attribute__((ext_vector_type(4)))  __bf16 bf16x4;
typedef __attribute__((ext_vector_type(8)))  float  v8f;
typedef __attribute__((ext_vector_type(4)))  unsigned int v4u;
typedef __attribute__((ext_vector_type(8)))  int v8i;
typedef __attribute__((ext_vector_type(4)))  int v4i;

#define NNODE 8192
#define DDIM  256
#define BM    64
#define BK    32
#define LDSPAD 8
#define MIN_NORM 1e-15f
#define MAXNORM  (1.0f - 1e-5f)

#if __has_builtin(__builtin_amdgcn_tensor_load_to_lds) && \
    __has_builtin(__builtin_amdgcn_s_wait_tensorcnt)
#define USE_TDM 1
#else
#define USE_TDM 0
#endif

__device__ __forceinline__ float artanh_(float v) {
    v = fminf(fmaxf(v, -1.0f + 1e-7f), 1.0f - 1e-7f);
    return 0.5f * (log1pf(v) - log1pf(-v));
}

// ---------------- prep: per-node norm of x (one wave32 per node) -------------
__global__ __launch_bounds__(256)
void prep_xn_kernel(const float* __restrict__ x, float* __restrict__ xn)
{
    int node = (blockIdx.x * blockDim.x + threadIdx.x) >> 5;
    int lane = threadIdx.x & 31;
    if (node >= NNODE) return;
    const float* row = x + (size_t)node * DDIM;
    float s = 0.0f;
    #pragma unroll
    for (int d = 0; d < DDIM / 32; ++d) {
        float v = row[lane + 32 * d];
        s += v * v;
    }
    #pragma unroll
    for (int m = 16; m; m >>= 1) s += __shfl_xor(s, m, 32);
    if (lane == 0) xn[node] = fmaxf(sqrtf(s), MIN_NORM);
}

// ---------------- prep: hyp_bias = proj(expmap0(bias)), y2 = ||hb||^2 --------
__global__ __launch_bounds__(256)
void prep_bias_kernel(const float* __restrict__ bias,
                      float* __restrict__ hb, float* __restrict__ y2out)
{
    __shared__ float red[DDIM];
    int t = threadIdx.x;
    float b = bias[t];
    red[t] = b * b;
    __syncthreads();
    for (int s = DDIM / 2; s; s >>= 1) {
        if (t < s) red[t] += red[t + s];
        __syncthreads();
    }
    float bn = fmaxf(sqrtf(red[0]), MIN_NORM);
    float tb = fminf(tanhf(bn), MAXNORM);   // expmap0 norm, then proj clamp
    hb[t] = (tb / bn) * b;
    if (t == 0) *y2out = tb * tb;
}

// ---------------- stage 1: HypLinear fused (GEMM x@W^T + hyperbolic chain) ---
__global__ __launch_bounds__(256)
void hyp_linear_kernel(const float* __restrict__ x,
                       const float* __restrict__ w,
                       const float* __restrict__ xn,
                       const float* __restrict__ hb,
                       const float* __restrict__ y2p,
                       bf16* __restrict__ xtT)
{
    __shared__ bf16  sA[BM][BK + LDSPAD];
    __shared__ bf16  sB[DDIM][BK + LDSPAD];
    __shared__ float redS[BM];
    __shared__ float redD[BM];
    __shared__ float shb[DDIM];

    const int tid    = threadIdx.x;
    const int wave   = tid >> 5;
    const int lane   = tid & 31;
    const int l15    = lane & 15;
    const int hi     = lane >> 4;
    const int mstrip = (wave & 3) * 16;
    const int nbase  = (wave >> 2) * 128;
    const int m0     = blockIdx.x * BM;

    shb[tid] = hb[tid];
    const float y2 = *y2p;

    v8f acc[8];
    #pragma unroll
    for (int t = 0; t < 8; ++t)
        #pragma unroll
        for (int i = 0; i < 8; ++i) acc[t][i] = 0.0f;

    for (int k0 = 0; k0 < DDIM; k0 += BK) {
        for (int i = tid; i < (BM * BK) / 4; i += 256) {
            int r = i >> 3, c4 = (i & 7) * 4;
            float4 v = *(const float4*)(x + (size_t)(m0 + r) * DDIM + k0 + c4);
            bf16x4 q = { (bf16)v.x, (bf16)v.y, (bf16)v.z, (bf16)v.w };
            *(bf16x4*)&sA[r][c4] = q;
        }
        for (int i = tid; i < (DDIM * BK) / 4; i += 256) {
            int r = i >> 3, c4 = (i & 7) * 4;
            float4 v = *(const float4*)(w + (size_t)r * DDIM + k0 + c4);
            bf16x4 q = { (bf16)v.x, (bf16)v.y, (bf16)v.z, (bf16)v.w };
            *(bf16x4*)&sB[r][c4] = q;
        }
        __syncthreads();

        const int arow = mstrip + l15;
        const int akb  = hi * 8;
        bf16x8 alo = *(const bf16x8*)&sA[arow][akb];
        bf16x8 ahi = *(const bf16x8*)&sA[arow][akb + 16];
        v16bf afrag;
        #pragma unroll
        for (int i = 0; i < 8; ++i) { afrag[i] = alo[i]; afrag[i + 8] = ahi[i]; }

        // group all B-fragment loads, then the dense WMMA chain
        const int bkb = hi * 8;
        v16bf bfr[8];
        #pragma unroll
        for (int t = 0; t < 8; ++t) {
            int n = nbase + t * 16 + l15;
            bf16x8 blo = *(const bf16x8*)&sB[n][bkb];
            bf16x8 bhi = *(const bf16x8*)&sB[n][bkb + 16];
            #pragma unroll
            for (int i = 0; i < 8; ++i) { bfr[t][i] = blo[i]; bfr[t][i + 8] = bhi[i]; }
        }
        #pragma unroll
        for (int t = 0; t < 8; ++t)
            acc[t] = __builtin_amdgcn_wmma_f32_16x16x32_bf16(
                false, afrag, false, bfr[t], (short)0, acc[t], false, false);
        __syncthreads();
    }

    // ---- epilogue: per-row Sum(mx^2) and <mx, hb> -------------------------
    if (tid < BM) { redS[tid] = 0.0f; redD[tid] = 0.0f; }
    __syncthreads();

    float rs[8], rd[8];
    #pragma unroll
    for (int j = 0; j < 8; ++j) { rs[j] = 0.0f; rd[j] = 0.0f; }
    #pragma unroll
    for (int t = 0; t < 8; ++t) {
        int n = nbase + t * 16 + l15;
        float hbv = shb[n];
        #pragma unroll
        for (int j = 0; j < 8; ++j) {
            float v = acc[t][j];
            rs[j] += v * v;
            rd[j] += v * hbv;
        }
    }
    #pragma unroll
    for (int m = 1; m < 16; m <<= 1) {
        #pragma unroll
        for (int j = 0; j < 8; ++j) {
            rs[j] += __shfl_xor(rs[j], m, 32);
            rd[j] += __shfl_xor(rd[j], m, 32);
        }
    }
    if (l15 == 0) {
        #pragma unroll
        for (int j = 0; j < 8; ++j) {
            int row = mstrip + hi * 8 + j;
            atomicAdd(&redS[row], rs[j]);
            atomicAdd(&redD[row], rd[j]);
        }
    }
    __syncthreads();

    float theta[8], phi[8];
    #pragma unroll
    for (int j = 0; j < 8; ++j) {
        int row   = mstrip + hi * 8 + j;
        float S1  = redS[row];
        float dot = redD[row];
        float xnv = xn[m0 + row];
        float mxn = fmaxf(sqrtf(S1), MIN_NORM);
        float t1  = tanhf(mxn / xnv * artanh_(xnv));
        float tc  = fminf(t1, MAXNORM);
        float s   = tc / mxn;
        float x2  = tc * tc;
        float xy  = s * dot;
        float den = fmaxf(1.0f + 2.0f * xy + x2 * y2, MIN_NORM);
        float al  = (1.0f + 2.0f * xy + y2) / den;
        float be  = (1.0f - x2) / den;
        float A1  = al * s;
        float h2q = al * al * x2 + 2.0f * al * be * xy + be * be * y2;
        float h2n = fmaxf(sqrtf(h2q), MIN_NORM);
        float gam = (h2n > MAXNORM) ? (MAXNORM / h2n) : 1.0f;
        float pn  = fmaxf(fminf(h2n, MAXNORM), MIN_NORM);
        float kap = artanh_(pn) / pn;
        theta[j] = kap * gam * A1;
        phi[j]   = kap * gam * be;
    }

    const int mbase = m0 + mstrip + hi * 8;
    #pragma unroll
    for (int t = 0; t < 8; ++t) {
        int n = nbase + t * 16 + l15;
        float hbv = shb[n];
        bf16x8 ov;
        #pragma unroll
        for (int j = 0; j < 8; ++j)
            ov[j] = (bf16)(theta[j] * acc[t][j] + phi[j] * hbv);
        *(bf16x8*)&xtT[(size_t)n * NNODE + mbase] = ov;
    }
}

// ---------------- stage 2: HypAgg + HypAct fused (adj @ Xt, 34 GFLOP) --------
__global__ __launch_bounds__(256)
void hyp_agg_kernel(const float* __restrict__ adj,
                    const bf16* __restrict__ xtT,
                    float* __restrict__ out)
{
    __shared__ bf16  sA[2][BM][BK + LDSPAD];
    __shared__ bf16  sB[2][DDIM][BK + LDSPAD];
    __shared__ float redS[BM];
    __shared__ float redP[BM];

    const int tid    = threadIdx.x;
    const int wave   = tid >> 5;
    const int lane   = tid & 31;
    const int l15    = lane & 15;
    const int hi     = lane >> 4;
    const int mstrip = (wave & 3) * 16;
    const int nbase  = (wave >> 2) * 128;
    const int m0     = blockIdx.x * BM;
    const int ar     = tid >> 3;          // adj stage row (2 float4/thread)
    const int ac     = (tid & 7) * 4;     // adj stage col

    v8f acc[8];
    #pragma unroll
    for (int t = 0; t < 8; ++t)
        #pragma unroll
        for (int i = 0; i < 8; ++i) acc[t][i] = 0.0f;

#if USE_TDM
    // Tensor Data Mover descriptor issue: 2D tile 256 rows x 64B of bf16 XtT,
    // tensor_dim0_stride = 8192 elems, LDS pad 4 DWORDs every 16 DWORDs
    // (reproduces the +8 bf16 row padding of sB). ISA cdna5 8.3-8.6.
    auto tdm_issue = [&](const bf16* gtile, unsigned lds_off) {
        unsigned long long ga = (unsigned long long)(size_t)gtile;
        v4u g0;
        g0[0] = 1u;                                      // count=1, no gather
        g0[1] = lds_off;                                 // lds_addr (bytes)
        g0[2] = (unsigned)(ga & 0xFFFFFFFFu);            // global_addr[31:0]
        g0[3] = (unsigned)((ga >> 32) & 0x01FFFFFFu)     // global_addr[56:32]
              | (2u << 30);                              // type=2 (image)
        v8i g1;
        g1[0] = (int)((1u << 16)      // data_size = 2 bytes
                    | (1u << 20)      // pad_enable
                    | (3u << 22)      // pad_interval: 16 DWORDs
                    | (3u << 25));    // pad_amount:   4 DWORDs
        g1[1] = (int)(((unsigned)NNODE & 0xFFFFu) << 16);        // tensor_dim0 lo
        g1[2] = (int)((((unsigned)NNODE >> 16) & 0xFFFFu)        // tensor_dim0 hi
                    | ((unsigned)DDIM << 16));                   // tensor_dim1 lo
        g1[3] = (int)((unsigned)BK << 16);                       // tile_dim0
        g1[4] = (int)DDIM;                                       // tile_dim1
        g1[5] = (int)NNODE;                                      // dim0_stride lo
        g1[6] = 0;
        g1[7] = 0;
        v4i gz4 = { 0, 0, 0, 0 };
        v8i gz8 = { 0, 0, 0, 0, 0, 0, 0, 0 };
        // 6-arg toolchain variant: (g0, g1, g2, g3, g4, cpol)
        __builtin_amdgcn_tensor_load_to_lds(g0, g1, gz4, gz4, gz8, 0);
    };
#endif

    float4 a0, a1;
#if !USE_TDM
    bf16x8 br[4];
#endif

    // ---- pipeline prologue: tile 0 into buffer 0 --------------------------
    {
#if USE_TDM
        if (wave == 0) tdm_issue(xtT, (unsigned)(size_t)&sB[0][0][0]);
#else
        #pragma unroll
        for (int j = 0; j < 4; ++j) {
            int i = tid + j * 256, r = i >> 2, c8 = (i & 3) * 8;
            br[j] = *(const bf16x8*)(xtT + (size_t)r * NNODE + c8);
        }
#endif
        a0 = *(const float4*)(adj + (size_t)(m0 + ar) * NNODE + ac);
        a1 = *(const float4*)(adj + (size_t)(m0 + ar + 32) * NNODE + ac);
        bf16x4 q0 = { (bf16)a0.x, (bf16)a0.y, (bf16)a0.z, (bf16)a0.w };
        bf16x4 q1 = { (bf16)a1.x, (bf16)a1.y, (bf16)a1.z, (bf16)a1.w };
        *(bf16x4*)&sA[0][ar][ac]      = q0;
        *(bf16x4*)&sA[0][ar + 32][ac] = q1;
#if !USE_TDM
        #pragma unroll
        for (int j = 0; j < 4; ++j) {
            int i = tid + j * 256, r = i >> 2, c8 = (i & 3) * 8;
            *(bf16x8*)&sB[0][r][c8] = br[j];
        }
#endif
    }

    const int T = NNODE / BK;   // 256 K-tiles
    for (int t = 0; t < T; ++t) {
        const int cur = t & 1;
        const int nxt = cur ^ 1;
        const bool has_next = (t + 1 < T);
        const int kn = (t + 1) * BK;

        if (has_next) {
#if USE_TDM
            if (wave == 0) tdm_issue(xtT + kn, (unsigned)(size_t)&sB[nxt][0][0]);
#else
            #pragma unroll
            for (int j = 0; j < 4; ++j) {
                int i = tid + j * 256, r = i >> 2, c8 = (i & 3) * 8;
                br[j] = *(const bf16x8*)(xtT + (size_t)r * NNODE + kn + c8);
            }
#endif
            const float* p0 = adj + (size_t)(m0 + ar) * NNODE + kn + ac;
            const float* p1 = adj + (size_t)(m0 + ar + 32) * NNODE + kn + ac;
            a0 = *(const float4*)p0;
            a1 = *(const float4*)p1;
            if (kn + BK < NNODE) __builtin_prefetch(p0 + BK, 0, 3);
        }
#if USE_TDM
        if (wave == 0) {
            if (has_next) __builtin_amdgcn_s_wait_tensorcnt(1);
            else          __builtin_amdgcn_s_wait_tensorcnt(0);
        }
#endif
        __syncthreads();   // tile `cur` (A stores + TDM/B stores) now visible

        const int arow = mstrip + l15;
        const int akb  = hi * 8;
        bf16x8 alo = *(const bf16x8*)&sA[cur][arow][akb];
        bf16x8 ahi = *(const bf16x8*)&sA[cur][arow][akb + 16];
        v16bf afrag;
        #pragma unroll
        for (int i = 0; i < 8; ++i) { afrag[i] = alo[i]; afrag[i + 8] = ahi[i]; }

        const int bkb = hi * 8;
        v16bf bfr[8];
        #pragma unroll
        for (int tt = 0; tt < 8; ++tt) {
            int n = nbase + tt * 16 + l15;
            bf16x8 blo = *(const bf16x8*)&sB[cur][n][bkb];
            bf16x8 bhi = *(const bf16x8*)&sB[cur][n][bkb + 16];
            #pragma unroll
            for (int i = 0; i < 8; ++i) { bfr[tt][i] = blo[i]; bfr[tt][i + 8] = bhi[i]; }
        }
        #pragma unroll
        for (int tt = 0; tt < 8; ++tt)
            acc[tt] = __builtin_amdgcn_wmma_f32_16x16x32_bf16(
                false, afrag, false, bfr[tt], (short)0, acc[tt], false, false);

        if (has_next) {
            bf16x4 q0 = { (bf16)a0.x, (bf16)a0.y, (bf16)a0.z, (bf16)a0.w };
            bf16x4 q1 = { (bf16)a1.x, (bf16)a1.y, (bf16)a1.z, (bf16)a1.w };
            *(bf16x4*)&sA[nxt][ar][ac]      = q0;
            *(bf16x4*)&sA[nxt][ar + 32][ac] = q1;
#if !USE_TDM
            #pragma unroll
            for (int j = 0; j < 4; ++j) {
                int i = tid + j * 256, r = i >> 2, c8 = (i & 3) * 8;
                *(bf16x8*)&sB[nxt][r][c8] = br[j];
            }
#endif
        }
        __syncthreads();   // everyone done reading `cur` before it is re-filled
    }

    // ---- epilogue: Sum(u^2) and Sum(relu(u)^2) per row --------------------
    if (tid < BM) { redS[tid] = 0.0f; redP[tid] = 0.0f; }
    __syncthreads();

    float rs[8], rp[8];
    #pragma unroll
    for (int j = 0; j < 8; ++j) { rs[j] = 0.0f; rp[j] = 0.0f; }
    #pragma unroll
    for (int t = 0; t < 8; ++t)
        #pragma unroll
        for (int j = 0; j < 8; ++j) {
            float v = acc[t][j];
            rs[j] += v * v;
            float r_ = fmaxf(v, 0.0f);
            rp[j] += r_ * r_;
        }
    #pragma unroll
    for (int m = 1; m < 16; m <<= 1) {
        #pragma unroll
        for (int j = 0; j < 8; ++j) {
            rs[j] += __shfl_xor(rs[j], m, 32);
            rp[j] += __shfl_xor(rp[j], m, 32);
        }
    }
    if (l15 == 0) {
        #pragma unroll
        for (int j = 0; j < 8; ++j) {
            int row = mstrip + hi * 8 + j;
            atomicAdd(&redS[row], rs[j]);
            atomicAdd(&redP[row], rp[j]);
        }
    }
    __syncthreads();

    float lam[8];
    #pragma unroll
    for (int j = 0; j < 8; ++j) {
        int row  = mstrip + hi * 8 + j;
        float S1 = redS[row];
        float S2 = redP[row];
        float un = fmaxf(sqrtf(S1), MIN_NORM);
        float hn = fminf(tanhf(un), MAXNORM);
        float eta = hn / un;
        float pn = fmaxf(hn, MIN_NORM);
        float kap = artanh_(pn) / pn * eta;
        float rn = fmaxf(kap * sqrtf(S2), MIN_NORM);
        float zeta = fminf(tanhf(rn), MAXNORM) / rn;
        lam[j] = zeta * kap;
    }

    #pragma unroll
    for (int t = 0; t < 8; ++t) {
        int n = nbase + t * 16 + l15;
        #pragma unroll
        for (int j = 0; j < 8; ++j) {
            int row = mstrip + hi * 8 + j;
            out[(size_t)(m0 + row) * DDIM + n] = lam[j] * fmaxf(acc[t][j], 0.0f);
        }
    }
}

// ---------------------------------------------------------------------------
extern "C" void kernel_launch(void* const* d_in, const int* in_sizes, int n_in,
                              void* d_out, int out_size, void* d_ws, size_t ws_size,
                              hipStream_t stream)
{
    (void)in_sizes; (void)n_in; (void)out_size; (void)ws_size;
    const float* x    = (const float*)d_in[0];   // [8192, 256]
    const float* adj  = (const float*)d_in[1];   // [8192, 8192]
    const float* w    = (const float*)d_in[2];   // [256, 256]
    const float* bias = (const float*)d_in[3];   // [256]
    float* out = (float*)d_out;                  // [8192, 256]

    bf16*  xtT = (bf16*)d_ws;                                   // 4 MB
    float* xn  = (float*)((char*)d_ws + (size_t)DDIM * NNODE * sizeof(bf16));
    float* hb  = xn + NNODE;
    float* y2  = hb + DDIM;

    prep_xn_kernel   <<<NNODE / 8, 256, 0, stream>>>(x, xn);
    prep_bias_kernel <<<1, DDIM, 0, stream>>>(bias, hb, y2);
    hyp_linear_kernel<<<NNODE / BM, 256, 0, stream>>>(x, w, xn, hb, y2, xtT);
    hyp_agg_kernel   <<<NNODE / BM, 256, 0, stream>>>(adj, xtT, out);
}
